// WindowAttentionLayer_28819230556274
// MI455X (gfx1250) — matrix operand order repaired and checked by reading, not compiled, not run on hardware
//
#include <hip/hip_runtime.h>
#include <hip/hip_bf16.h>
#include <math.h>
#include <stdint.h>

typedef __attribute__((ext_vector_type(16))) _Float16 v16h;
typedef __attribute__((ext_vector_type(8)))  float    v8f;
typedef __attribute__((ext_vector_type(4)))  unsigned int v4u;
typedef __attribute__((ext_vector_type(8)))  int      v8i;
typedef __attribute__((ext_vector_type(4)))  int      v4i;

#define USE_TDM 1

#define NTOK   64
#define CDIM   192
#define NHEADS 6
#define DH     32

// padded LDS strides (elements); all row starts 16B aligned
#define XS_STR   200   // f16, 400B/row
#define QKV_STR  584   // f16, 1168B/row
#define ATT_STR  68    // f32, 272B/row
#define OUT_STR  200   // f16, 400B/row

#define QKVW_ELEMS (576 * 192)   // 110592
#define PROJW_ELEMS (192 * 192)  //  36864
#define WS_F16_BYTES ((QKVW_ELEMS + PROJW_ELEMS) * 2)

__device__ __forceinline__ v8f wmma16(v16h a, v16h b, v8f c) {
    return __builtin_amdgcn_wmma_f32_16x16x32_f16(
        false, a, false, b, (short)0, c, false, false);
}

// A fragment (16x32 f16): lane m = l&15; K run = (l>>4)*16 (16 contiguous).
__device__ __forceinline__ v16h load_a_lds(const _Float16* base, int stride,
                                           int m0, int k0, int lane) {
    v16h a;
    const _Float16* r = base + (m0 + (lane & 15)) * stride;
    int kA = k0 + ((lane >> 4) << 4);
#pragma unroll
    for (int j = 0; j < 8; ++j) {
        a[j]     = r[kA + j];
        a[8 + j] = r[kA + 8 + j];
    }
    return a;
}

// A fragment sourced from f32 LDS (attention probabilities), converted to f16.
__device__ __forceinline__ v16h load_a_lds_f32(const float* base, int stride,
                                               int m0, int k0, int lane) {
    v16h a;
    const float* r = base + (m0 + (lane & 15)) * stride;
    int kA = k0 + ((lane >> 4) << 4);
#pragma unroll
    for (int j = 0; j < 8; ++j) {
        a[j]     = (_Float16)r[kA + j];
        a[8 + j] = (_Float16)r[kA + 8 + j];
    }
    return a;
}

// B fragment (32x16 f16): lane col n = l&15, K run = (l>>4)*16 (16 contiguous).
// f32 weight fallback: B[k,n] = W[(n0+n)*ldw + k0+k], convert on the fly.
__device__ __forceinline__ v16h load_b_weight_f32(const float* __restrict__ W, int ldw,
                                                  int n0, int k0, int lane) {
    v16h b;
    const float* r = W + (n0 + (lane & 15)) * ldw + k0 + ((lane >> 4) << 4);
#pragma unroll
    for (int j = 0; j < 16; ++j) b[j] = (_Float16)r[j];
    return b;
}

// pre-converted f16 weights: 16 contiguous f16 = one 32B run per lane.
__device__ __forceinline__ v16h load_b_weight_f16(const _Float16* __restrict__ W, int ldw,
                                                  int n0, int k0, int lane) {
    v16h b;
    const _Float16* r = W + (n0 + (lane & 15)) * ldw + k0 + ((lane >> 4) << 4);
#pragma unroll
    for (int j = 0; j < 16; ++j) b[j] = r[j];
    return b;
}

// B fragment, k contiguous within an LDS row: B[k,n] = base[(n0+n)*str + colbase + k]
__device__ __forceinline__ v16h load_b_rowk(const _Float16* base, int stride,
                                            int n0, int colbase, int lane) {
    v16h b;
    const _Float16* r = base + (n0 + (lane & 15)) * stride + colbase + ((lane >> 4) << 4);
#pragma unroll
    for (int j = 0; j < 16; ++j) b[j] = r[j];
    return b;
}

// B fragment, k strides across LDS rows: B[k,n] = base[(k0+k)*str + col0 + n]
__device__ __forceinline__ v16h load_b_colk(const _Float16* base, int stride,
                                            int k0, int col0, int lane) {
    v16h b;
    int c  = col0 + (lane & 15);
    int ks = k0 + ((lane >> 4) << 4);
#pragma unroll
    for (int j = 0; j < 16; ++j) b[j] = base[(ks + j) * stride + c];
    return b;
}

// one-shot weight conversion into workspace (f32 -> f16)
__global__ __launch_bounds__(256)
void cvt_weights_kernel(const float* __restrict__ qkv_w,
                        const float* __restrict__ proj_w,
                        _Float16* __restrict__ dst) {
    int i = blockIdx.x * 256 + threadIdx.x;
    if (i < QKVW_ELEMS)                       dst[i] = (_Float16)qkv_w[i];
    else if (i < QKVW_ELEMS + PROJW_ELEMS)    dst[i] = (_Float16)proj_w[i - QKVW_ELEMS];
}

template <bool WF16>
__global__ __launch_bounds__(256)
void win_attn_kernel(const float* __restrict__ x,
                     const float* __restrict__ mask,
                     const void*  __restrict__ qkv_wv,
                     const float* __restrict__ qkv_b,
                     const void*  __restrict__ proj_wv,
                     const float* __restrict__ proj_b,
                     const float* __restrict__ bias_table,
                     float* __restrict__ out) {
    __shared__ _Float16 xs   [NTOK * XS_STR];
    __shared__ _Float16 qkvs [NTOK * QKV_STR];
    __shared__ float    atts [NTOK * ATT_STR];
    __shared__ _Float16 outh [NTOK * OUT_STR];
    __shared__ float    masks[NTOK * NTOK];    // staged by TDM

    const int w    = blockIdx.x;
    const int tid  = threadIdx.x;
    const int wave = tid >> 5;
    const int lane = tid & 31;
    const float scale = 0.17677669529663687f;   // dh^-0.5

    const float* maskw = mask + (size_t)w * NTOK * NTOK;

#if USE_TDM
    // ---- Kick off TDM copy of this window's mask tile (64x64 f32) into LDS.
    // Descriptor per CDNA5 ISA; all fields uniform.
    if (wave == 0) {
        uint64_t ga  = (uint64_t)(uintptr_t)maskw;
        uint32_t lds = (uint32_t)(uintptr_t)(&masks[0]);
        v4u g0;
        g0[0] = 1u;                                     // count=1, user desc
        g0[1] = lds;                                    // lds_addr (bytes)
        g0[2] = (uint32_t)ga;                           // global_addr[31:0]
        g0[3] = (uint32_t)((ga >> 32) & 0x1FFFFFFu)     // global_addr[56:32]
              | (2u << 30);                             // type=2 ("image")
        v8i g1;
        g1[0] = (int)(2u << 16);        // wg_mask=0, data_size=4B, no flags
        g1[1] = (int)(64u << 16);       // tensor_dim0 = 64 (lo16 @ bits63:48)
        g1[2] = (int)(64u << 16);       // tensor_dim0 hi=0 | tensor_dim1=64
        g1[3] = (int)(64u << 16);       // tensor_dim1 hi=0 | tile_dim0=64
        g1[4] = (int)64u;               // tile_dim1=64, tile_dim2=0
        g1[5] = (int)64u;               // tensor_dim0_stride = 64 (lo32)
        g1[6] = 0;                      // stride hi, tensor_dim1_stride lo
        g1[7] = 0;
        v4i gz4 = {0, 0, 0, 0};         // groups 2/3 unused (2D tensor)
        v8i gz8 = {0, 0, 0, 0, 0, 0, 0, 0};
        __builtin_amdgcn_tensor_load_to_lds(g0, g1, gz4, gz4, gz8, 0);
    }
#else
    for (int i = tid; i < NTOK * NTOK; i += 256) masks[i] = maskw[i];
#endif

    // ---- Stage 0: stage x[w] (64x192 f32) into LDS as f16 -------------------
    const float* xw = x + (size_t)w * NTOK * CDIM;
    for (int i = tid; i < NTOK * CDIM; i += 256) {
        int r = i / CDIM, c = i - r * CDIM;
        xs[r * XS_STR + c] = (_Float16)xw[i];
    }
    __syncthreads();

    const _Float16* qkv_wh = (const _Float16*)qkv_wv;
    const float*    qkv_wf = (const float*)qkv_wv;
    const _Float16* proj_wh = (const _Float16*)proj_wv;
    const float*    proj_wf = (const float*)proj_wv;

    // ---- Stage 1: QKV GEMM  [64,192] x [192,576] -> qkvs (q pre-scaled) -----
    for (int t = wave; t < 4 * 36; t += 8) {
        int m0 = (t / 36) * 16;
        int n0 = (t % 36) * 16;
        v8f acc = {};
#pragma unroll
        for (int kk = 0; kk < 6; ++kk) {
            v16h a = load_a_lds(xs, XS_STR, m0, kk * 32, lane);
            v16h b = WF16 ? load_b_weight_f16(qkv_wh, CDIM, n0, kk * 32, lane)
                          : load_b_weight_f32(qkv_wf, CDIM, n0, kk * 32, lane);
            acc = wmma16(a, b, acc);
        }
        int col = n0 + (lane & 15);
        float bias = qkv_b[col];
        float s = (col < CDIM) ? scale : 1.0f;   // scale q only
#pragma unroll
        for (int r = 0; r < 8; ++r) {
            int row = m0 + r + ((lane >> 4) << 3);
            qkvs[row * QKV_STR + col] = (_Float16)((acc[r] + bias) * s);
        }
    }
#if USE_TDM
    if (wave == 0) __builtin_amdgcn_s_wait_tensorcnt(0);  // mask tile resident
#endif
    __syncthreads();

    // ---- Stage 2: per-head attention ---------------------------------------
    for (int h = 0; h < NHEADS; ++h) {
        // 2a: logits = q @ k^T : 4x4 tiles, K=32 -> single WMMA per tile
        for (int t = wave; t < 16; t += 8) {
            int m0 = (t >> 2) * 16;
            int n0 = (t & 3) * 16;
            v16h a = load_a_lds(qkvs, QKV_STR, m0, h * DH, lane);           // q
            v16h b = load_b_rowk(qkvs, QKV_STR, n0, CDIM + h * DH, lane);   // k
            v8f acc = {};
            acc = wmma16(a, b, acc);
#pragma unroll
            for (int r = 0; r < 8; ++r) {
                int row = m0 + r + ((lane >> 4) << 3);
                atts[row * ATT_STR + n0 + (lane & 15)] = acc[r];
            }
        }
        __syncthreads();

        // 2b: softmax row-wise with relative-position bias + LDS-resident mask
        if (tid < NTOK) {
            int m = tid;
            int mr = m >> 3, mc = m & 7;
            float mx = -1e30f;
            for (int n = 0; n < NTOK; ++n) {
                int nr = n >> 3, nc = n & 7;
                int rpi = (mr - nr + 7) * 15 + (mc - nc + 7);
                float v = atts[m * ATT_STR + n] + bias_table[rpi * NHEADS + h]
                        + masks[m * NTOK + n];
                mx = fmaxf(mx, v);
            }
            float sum = 0.0f;
            for (int n = 0; n < NTOK; ++n) {
                int nr = n >> 3, nc = n & 7;
                int rpi = (mr - nr + 7) * 15 + (mc - nc + 7);
                float v = atts[m * ATT_STR + n] + bias_table[rpi * NHEADS + h]
                        + masks[m * NTOK + n];
                float e = __expf(v - mx);
                atts[m * ATT_STR + n] = e;
                sum += e;
            }
            float inv = 1.0f / sum;
            for (int n = 0; n < NTOK; ++n) atts[m * ATT_STR + n] *= inv;
        }
        __syncthreads();

        // 2c: out_h = attn @ v : 4 m-tiles x 2 n-tiles = 8 tiles (1/wave), K=64
        {
            int m0 = (wave >> 1) * 16;
            int n0 = (wave & 1) * 16;
            v8f acc = {};
#pragma unroll
            for (int kk = 0; kk < 2; ++kk) {
                v16h a = load_a_lds_f32(atts, ATT_STR, m0, kk * 32, lane);
                v16h b = load_b_colk(qkvs, QKV_STR, kk * 32,
                                     2 * CDIM + h * DH + n0, lane);          // v
                acc = wmma16(a, b, acc);
            }
#pragma unroll
            for (int r = 0; r < 8; ++r) {
                int row = m0 + r + ((lane >> 4) << 3);
                outh[row * OUT_STR + h * DH + n0 + (lane & 15)] = (_Float16)acc[r];
            }
        }
        __syncthreads();   // protect atts/outh across head iterations
    }

    // ---- Stage 3: output projection  [64,192] x [192,192] + bias -----------
    float* yw = out + (size_t)w * NTOK * CDIM;
    for (int t = wave; t < 4 * 12; t += 8) {
        int m0 = (t / 12) * 16;
        int n0 = (t % 12) * 16;
        v8f acc = {};
#pragma unroll
        for (int kk = 0; kk < 6; ++kk) {
            v16h a = load_a_lds(outh, OUT_STR, m0, kk * 32, lane);
            v16h b = WF16 ? load_b_weight_f16(proj_wh, CDIM, n0, kk * 32, lane)
                          : load_b_weight_f32(proj_wf, CDIM, n0, kk * 32, lane);
            acc = wmma16(a, b, acc);
        }
        int col = n0 + (lane & 15);
        float pb = proj_b[col];
#pragma unroll
        for (int r = 0; r < 8; ++r) {
            int row = m0 + r + ((lane >> 4) << 3);
            yw[row * CDIM + col] = acc[r] + pb;
        }
    }
}

extern "C" void kernel_launch(void* const* d_in, const int* in_sizes, int n_in,
                              void* d_out, int out_size, void* d_ws, size_t ws_size,
                              hipStream_t stream) {
    const float* x          = (const float*)d_in[0];
    const float* mask       = (const float*)d_in[1];
    const float* qkv_w      = (const float*)d_in[2];
    const float* qkv_b      = (const float*)d_in[3];
    const float* proj_w     = (const float*)d_in[4];
    const float* proj_b     = (const float*)d_in[5];
    const float* bias_table = (const float*)d_in[6];
    float* out              = (float*)d_out;

    const int nW = in_sizes[0] / (NTOK * CDIM);   // 4096

    if (ws_size >= (size_t)WS_F16_BYTES) {
        _Float16* wsh = (_Float16*)d_ws;
        int total = QKVW_ELEMS + PROJW_ELEMS;
        cvt_weights_kernel<<<(total + 255) / 256, 256, 0, stream>>>(qkv_w, proj_w, wsh);
        win_attn_kernel<true><<<nW, 256, 0, stream>>>(
            x, mask, wsh, qkv_b, wsh + QKVW_ELEMS, proj_b, bias_table, out);
    } else {
        win_attn_kernel<false><<<nW, 256, 0, stream>>>(
            x, mask, qkv_w, qkv_b, proj_w, proj_b, bias_table, out);
    }
}
